// NonLocalFusionBlock_4423816315249
// MI455X (gfx1250) — matrix operand order repaired and checked by reading, not compiled, not run on hardware
//
#include <hip/hip_runtime.h>

typedef __attribute__((ext_vector_type(16))) _Float16 v16h;
typedef __attribute__((ext_vector_type(8)))  float    v8f;
typedef __attribute__((ext_vector_type(4)))  int      vi4;

#define TILE_M 64
#define TILE_N 128
#define TILE_K 32
#define LDA    40
#define LDB    40

#if __has_builtin(__builtin_amdgcn_global_load_async_to_lds_b128) && \
    __has_builtin(__builtin_amdgcn_s_wait_asynccnt)
#define USE_ASYNC_LDS 1
#endif

#ifdef USE_ASYNC_LDS
typedef __attribute__((address_space(1))) vi4 as1_vi4;
typedef __attribute__((address_space(3))) vi4 as3_vi4;
__device__ __forceinline__ void async_copy16(const _Float16* g, _Float16* l) {
  // global_load_async_to_lds_b128 (ASYNCcnt-tracked)
  __builtin_amdgcn_global_load_async_to_lds_b128((as1_vi4*)g, (as3_vi4*)l, 0, 0);
}
#endif

__device__ __forceinline__ v8f wmma32(v16h a, v16h b, v8f c) {
  // v_wmma_f32_16x16x32_f16
  return __builtin_amdgcn_wmma_f32_16x16x32_f16(false, a, false, b, (short)0, c,
                                                false, false);
}

// A-fragment (16x32 f16): lane M = lane&15; halves 0..7 <- k = hi*8..hi*8+7,
// halves 8..15 <- k = 16+hi*8..16+hi*8+7   (hi = lane>=16)
__device__ __forceinline__ v16h frag_a(const _Float16* row, int hi) {
  v16h r;
  uint4* p = reinterpret_cast<uint4*>(&r);
  p[0] = *reinterpret_cast<const uint4*>(row + hi * 8);
  p[1] = *reinterpret_cast<const uint4*>(row + 16 + hi * 8);
  return r;
}

// B-fragment (32x16 f16): lane N = lane&15; halves 0..7 <- k = hi*8..hi*8+7,
// halves 8..15 <- k = 16+hi*8..16+hi*8+7
__device__ __forceinline__ v16h frag_b(const _Float16* row, int hi) {
  v16h r;
  uint4* p = reinterpret_cast<uint4*>(&r);
  p[0] = *reinterpret_cast<const uint4*>(row + hi * 8);
  p[1] = *reinterpret_cast<const uint4*>(row + 16 + hi * 8);
  return r;
}

// convert 8 consecutive f32 -> packed 8 x f16, one 16B LDS store
__device__ __forceinline__ void cvt_store8(_Float16* dst, float4 f0, float4 f1) {
  union { _Float16 h[8]; uint4 u; } t;
  t.h[0] = (_Float16)f0.x; t.h[1] = (_Float16)f0.y;
  t.h[2] = (_Float16)f0.z; t.h[3] = (_Float16)f0.w;
  t.h[4] = (_Float16)f1.x; t.h[5] = (_Float16)f1.y;
  t.h[6] = (_Float16)f1.z; t.h[7] = (_Float16)f1.w;
  *reinterpret_cast<uint4*>(dst) = t.u;
}

// ---------------------------------------------------------------------------
// 1x1 projection GEMM: acc[m,n] = sum_k W[m,k] * X[b,k,n] + bias[m]
// M = 128 channels, N = 4096 positions, K = Cin (256 or 512). Out is f16.
// TRANS==0: Out[b][n][m] (row per position);  TRANS==1: Out[b][m][n].
// ---------------------------------------------------------------------------
template <int TRANS>
__global__ __launch_bounds__(256) void proj_kernel(
    const float* __restrict__ X, const float* __restrict__ W,
    const float* __restrict__ bias, _Float16* __restrict__ Out, int Cin) {
  __shared__ _Float16 As[TILE_M * LDA];
  __shared__ _Float16 Bs[TILE_N * LDB];
  const int b  = blockIdx.z;
  const int n0 = blockIdx.x * TILE_N;
  const int m0 = blockIdx.y * TILE_M;
  const float* Xb = X + (size_t)b * Cin * 4096;
  const int tid = threadIdx.x;
  const int wid = tid >> 5, lane = tid & 31;
  const int wm = wid & 3, wn = wid >> 2;     // wave tile: 16 rows x 64 cols
  const int lo = lane & 15, hi = lane >> 4;

  v8f acc[4] = {};
  for (int k0 = 0; k0 < Cin; k0 += TILE_K) {
    __syncthreads();
    {  // A tile: 64x32, thread -> (m = tid>>2, 8 k's); b128 reads, b128 store
      int m = tid >> 2, k8 = (tid & 3) * 8;
      const float* wp = W + (size_t)(m0 + m) * Cin + k0 + k8;
      cvt_store8(As + m * LDA + k8, *(const float4*)wp, *(const float4*)(wp + 4));
    }
    {  // B tile: [n][k] <- X[k][n]; thread -> (k = tid>>3, 16 n's)
      int k = tid >> 3, n4 = (tid & 7) * 16;
      const float* src = Xb + (size_t)(k0 + k) * 4096 + n0 + n4;
#pragma unroll
      for (int q = 0; q < 4; ++q) {
        float4 f = *(const float4*)(src + q * 4);
        Bs[(n4 + q * 4 + 0) * LDB + k] = (_Float16)f.x;
        Bs[(n4 + q * 4 + 1) * LDB + k] = (_Float16)f.y;
        Bs[(n4 + q * 4 + 2) * LDB + k] = (_Float16)f.z;
        Bs[(n4 + q * 4 + 3) * LDB + k] = (_Float16)f.w;
      }
    }
    __syncthreads();
    v16h a = frag_a(As + (wm * 16 + lo) * LDA, hi);
#pragma unroll
    for (int j = 0; j < 4; ++j) {
      v16h bf = frag_b(Bs + (wn * 64 + j * 16 + lo) * LDB, hi);
      acc[j] = wmma32(a, bf, acc[j]);
    }
  }
#pragma unroll
  for (int j = 0; j < 4; ++j) {
    int n = n0 + wn * 64 + j * 16 + lo;
#pragma unroll
    for (int r = 0; r < 8; ++r) {
      int m = m0 + wm * 16 + hi * 8 + r;
      _Float16 v = (_Float16)(acc[j][r] + bias[m]);
      if (TRANS)
        Out[(size_t)b * 4096 * 128 + (size_t)m * 4096 + n] = v;
      else
        Out[(size_t)b * 4096 * 128 + (size_t)n * 128 + m] = v;
    }
  }
}

// ---------------------------------------------------------------------------
// Flash attention: S = Q K, online softmax over 4096 keys, O = P V.
// Qh/Kth: [B][4096][128]; VhT: [B][128][4096]; OT out: [B][128][4096] f16.
// Each wave: 16 query rows, full 128-wide output. Async-prefetched, double-
// buffered K/V tiles (ASYNCcnt), one barrier per key-tile.
// ---------------------------------------------------------------------------
#ifdef USE_ASYNC_LDS
#define NBUF 2
#else
#define NBUF 1
#endif

#ifdef USE_ASYNC_LDS
__device__ __forceinline__ void attn_issue_tile(const _Float16* Kb,
                                                const _Float16* Vb, int c0,
                                                _Float16* KsBuf,
                                                _Float16* VsBuf, int tid) {
  // K tile: 32 rows x 256B -> 512 x 16B chunks (2 per thread)
#pragma unroll
  for (int i = tid; i < 512; i += 256) {
    int rr = i >> 4, ch = i & 15;
    async_copy16(Kb + (size_t)(c0 + rr) * 128 + ch * 8, KsBuf + rr * 136 + ch * 8);
  }
  // V tile: 128 rows x 64B -> 512 x 16B chunks (2 per thread)
#pragma unroll
  for (int i = tid; i < 512; i += 256) {
    int cc = i >> 2, ch = i & 3;
    async_copy16(Vb + (size_t)cc * 4096 + c0 + ch * 8, VsBuf + cc * 40 + ch * 8);
  }
}
#endif

__global__ __launch_bounds__(256) void attn_kernel(
    const _Float16* __restrict__ Qh, const _Float16* __restrict__ Kth,
    const _Float16* __restrict__ VhT, _Float16* __restrict__ OT) {
  __shared__ _Float16 Ks[NBUF][32 * 136];    // [key][c]  (pad 8)
  __shared__ _Float16 Vst[NBUF][128 * 40];   // [c][key]  (pad 8)
  __shared__ _Float16 Ps[8 * 16 * 40];       // per-wave P tile [16 r][32 keys]
  const int b  = blockIdx.y;
  const int r0 = blockIdx.x * 128;
  const int tid = threadIdx.x;
  const int wid = tid >> 5, lane = tid & 31;
  const int lo = lane & 15, hi = lane >> 4;

  const _Float16* Qb = Qh + ((size_t)b * 4096 + r0 + wid * 16) * 128;
  v16h qa[4];
#pragma unroll
  for (int kb = 0; kb < 4; ++kb)
    qa[kb] = frag_a(Qb + (size_t)lo * 128 + kb * 32, hi);

  v8f o[8] = {};
  float mrow[8], lrow[8];
#pragma unroll
  for (int r = 0; r < 8; ++r) { mrow[r] = -1e30f; lrow[r] = 0.f; }

  const _Float16* Kb = Kth + (size_t)b * 4096 * 128;
  const _Float16* Vb = VhT + (size_t)b * 4096 * 128;
  _Float16* Pw = Ps + wid * 16 * 40;

#ifdef USE_ASYNC_LDS
  attn_issue_tile(Kb, Vb, 0, Ks[0], Vst[0], tid);   // prefetch first tile
#endif

  for (int c0 = 0, t = 0; c0 < 4096; c0 += 32, ++t) {
#ifdef USE_ASYNC_LDS
    const int cur = t & 1;
    __builtin_amdgcn_s_wait_asynccnt(0);  // this wave's tile-t chunks landed
    __syncthreads();                      // all waves' chunks landed; prev
                                          // buffer's readers are done
    if (c0 + 32 < 4096)                   // prefetch next tile into other buf
      attn_issue_tile(Kb, Vb, c0 + 32, Ks[cur ^ 1], Vst[cur ^ 1], tid);
    const _Float16* KsC = Ks[cur];
    const _Float16* VsC = Vst[cur];
#else
    __syncthreads();
    for (int i = tid; i < 32 * 128; i += 256) {
      int rr = i >> 7, cc = i & 127;   // coalesced along cc
      Ks[0][rr * 136 + cc] = Kb[(size_t)(c0 + rr) * 128 + cc];
    }
    for (int i = tid; i < 128 * 32; i += 256) {
      int cc = i >> 5, rr = i & 31;    // coalesced along rr
      Vst[0][cc * 40 + rr] = Vb[(size_t)cc * 4096 + c0 + rr];
    }
    __syncthreads();
    const _Float16* KsC = Ks[0];
    const _Float16* VsC = Vst[0];
#endif

    v8f s0 = {}, s1 = {};
#pragma unroll
    for (int kb = 0; kb < 4; ++kb) {
      s0 = wmma32(qa[kb], frag_b(KsC + lo * 136 + kb * 32, hi), s0);
      s1 = wmma32(qa[kb], frag_b(KsC + (16 + lo) * 136 + kb * 32, hi), s1);
    }

    float scale[8];
#pragma unroll
    for (int r = 0; r < 8; ++r) {
      float v = fmaxf(s0[r], s1[r]);
      v = fmaxf(v, __shfl_xor(v, 1, 16));
      v = fmaxf(v, __shfl_xor(v, 2, 16));
      v = fmaxf(v, __shfl_xor(v, 4, 16));
      v = fmaxf(v, __shfl_xor(v, 8, 16));
      float nm = fmaxf(mrow[r], v);
      scale[r] = __expf(mrow[r] - nm);
      mrow[r] = nm;
    }
#pragma unroll
    for (int r = 0; r < 8; ++r) {
      float p0 = __expf(s0[r] - mrow[r]);
      float p1 = __expf(s1[r] - mrow[r]);
      float rs = p0 + p1;
      rs += __shfl_xor(rs, 1, 16);
      rs += __shfl_xor(rs, 2, 16);
      rs += __shfl_xor(rs, 4, 16);
      rs += __shfl_xor(rs, 8, 16);
      lrow[r] = lrow[r] * scale[r] + rs;
      int row = hi * 8 + r;            // C-layout row of this lane/VGPR
      Pw[row * 40 + lo]      = (_Float16)p0;
      Pw[row * 40 + 16 + lo] = (_Float16)p1;
#pragma unroll
      for (int j = 0; j < 8; ++j) o[j][r] *= scale[r];
    }

    v16h pa = frag_a(Pw + lo * 40, hi);   // wave-private LDS RAW (DScnt wait)
#pragma unroll
    for (int j = 0; j < 8; ++j)
      o[j] = wmma32(pa, frag_b(VsC + (j * 16 + lo) * 40, hi), o[j]);
  }

  float inv[8];
#pragma unroll
  for (int r = 0; r < 8; ++r) inv[r] = 1.0f / lrow[r];
  _Float16* Ob = OT + (size_t)b * 128 * 4096;
  const int rowbase = r0 + wid * 16 + hi * 8;
#pragma unroll
  for (int j = 0; j < 8; ++j) {
    int c = j * 16 + lo;
#pragma unroll
    for (int r = 0; r < 8; ++r)
      Ob[(size_t)c * 4096 + rowbase + r] = (_Float16)(o[j][r] * inv[r]);
  }
}

// ---------------------------------------------------------------------------
// Output projection + residual: Xh[b,o,n] = f2d[b,o,n] + Wo[o,:]·attnT[:,n]+bo
// M = 256, N = 4096, K = 128. Stores f16 for the conv kernel.
// ---------------------------------------------------------------------------
__global__ __launch_bounds__(256) void wo_kernel(
    const _Float16* __restrict__ AT, const float* __restrict__ Wo,
    const float* __restrict__ bo, const float* __restrict__ f2d,
    _Float16* __restrict__ Xh) {
  __shared__ _Float16 As[TILE_M * LDA];
  __shared__ _Float16 Bs[TILE_N * LDB];
  const int b  = blockIdx.z;
  const int n0 = blockIdx.x * TILE_N;
  const int m0 = blockIdx.y * TILE_M;
  const _Float16* Ab = AT + (size_t)b * 128 * 4096;
  const int tid = threadIdx.x;
  const int wid = tid >> 5, lane = tid & 31;
  const int wm = wid & 3, wn = wid >> 2;
  const int lo = lane & 15, hi = lane >> 4;

  v8f acc[4] = {};
  for (int k0 = 0; k0 < 128; k0 += TILE_K) {
    __syncthreads();
    {  // A tile from Wo (f32)
      int m = tid >> 2, k8 = (tid & 3) * 8;
      const float* wp = Wo + (size_t)(m0 + m) * 128 + k0 + k8;
      cvt_store8(As + m * LDA + k8, *(const float4*)wp, *(const float4*)(wp + 4));
    }
    {  // B tile [n][k] <- attnT[k][n] (f16): thread -> (k, 16 n's)
      int k = tid >> 3, n4 = (tid & 7) * 16;
      const _Float16* src = Ab + (size_t)(k0 + k) * 4096 + n0 + n4;
      union { uint4 u; _Float16 h[8]; } a0, a1;
      a0.u = *(const uint4*)src;
      a1.u = *(const uint4*)(src + 8);
#pragma unroll
      for (int j = 0; j < 8; ++j) {
        Bs[(n4 + j) * LDB + k]     = a0.h[j];
        Bs[(n4 + 8 + j) * LDB + k] = a1.h[j];
      }
    }
    __syncthreads();
    v16h a = frag_a(As + (wm * 16 + lo) * LDA, hi);
#pragma unroll
    for (int j = 0; j < 4; ++j)
      acc[j] = wmma32(a, frag_b(Bs + (wn * 64 + j * 16 + lo) * LDB, hi), acc[j]);
  }
#pragma unroll
  for (int j = 0; j < 4; ++j) {
    int n = n0 + wn * 64 + j * 16 + lo;
#pragma unroll
    for (int r = 0; r < 8; ++r) {
      int m = m0 + wm * 16 + hi * 8 + r;
      float v = acc[j][r] + bo[m] + f2d[((size_t)b * 256 + m) * 4096 + n];
      Xh[((size_t)b * 256 + m) * 4096 + n] = (_Float16)v;
    }
  }
}

// ---------------------------------------------------------------------------
// 3x3 SAME conv as implicit GEMM: 9 taps of K=256 GEMM accumulation.
// Y[b,o,p] = bps[o] + sum_{tap,c} Wps[o,c,tap] * x[b,c,shift_tap(p)]
// ---------------------------------------------------------------------------
__global__ __launch_bounds__(256) void conv_kernel(
    const _Float16* __restrict__ Xh, const float* __restrict__ Wps,
    const float* __restrict__ bps, float* __restrict__ Y) {
  __shared__ _Float16 As[TILE_M * LDA];
  __shared__ _Float16 Bs[TILE_N * LDB];
  const int b  = blockIdx.z;
  const int n0 = blockIdx.x * TILE_N;   // 128 pixels = 2 image rows
  const int m0 = blockIdx.y * TILE_M;
  const int y0 = n0 >> 6;
  const int tid = threadIdx.x;
  const int wid = tid >> 5, lane = tid & 31;
  const int wm = wid & 3, wn = wid >> 2;
  const int lo = lane & 15, hi = lane >> 4;

  v8f acc[4] = {};
  for (int tap = 0; tap < 9; ++tap) {
    const int dy = tap / 3 - 1, dx = tap % 3 - 1;
    for (int k0 = 0; k0 < 256; k0 += TILE_K) {
      __syncthreads();
      {  // A tile: strided (x9) gather from Wps, packed b128 LDS store
        int m = tid >> 2, k8 = (tid & 3) * 8;
        const float* wp =
            Wps + (size_t)(m0 + m) * 2304 + (size_t)(k0 + k8) * 9 + tap;
        union { _Float16 h[8]; uint4 u; } t;
#pragma unroll
        for (int j = 0; j < 8; ++j) t.h[j] = (_Float16)wp[j * 9];
        *reinterpret_cast<uint4*>(As + m * LDA + k8) = t.u;
      }
      {  // B tile with spatial shift + zero pad
        int k = tid >> 3, n4 = (tid & 7) * 16;
#pragma unroll
        for (int j = 0; j < 16; ++j) {
          int n = n4 + j;
          int yy = y0 + (n >> 6) + dy;
          int xx = (n & 63) + dx;
          _Float16 v = (_Float16)0.f;
          if ((unsigned)yy < 64u && (unsigned)xx < 64u)
            v = Xh[((size_t)b * 256 + k0 + k) * 4096 + yy * 64 + xx];
          Bs[n * LDB + k] = v;
        }
      }
      __syncthreads();
      v16h a = frag_a(As + (wm * 16 + lo) * LDA, hi);
#pragma unroll
      for (int j = 0; j < 4; ++j)
        acc[j] =
            wmma32(a, frag_b(Bs + (wn * 64 + j * 16 + lo) * LDB, hi), acc[j]);
    }
  }
#pragma unroll
  for (int j = 0; j < 4; ++j) {
    int n = n0 + wn * 64 + j * 16 + lo;
#pragma unroll
    for (int r = 0; r < 8; ++r) {
      int m = m0 + wm * 16 + hi * 8 + r;
      Y[((size_t)b * 256 + m) * 4096 + n] = acc[j][r] + bps[m];
    }
  }
}

extern "C" void kernel_launch(void* const* d_in, const int* in_sizes, int n_in,
                              void* d_out, int out_size, void* d_ws,
                              size_t ws_size, hipStream_t stream) {
  (void)in_sizes; (void)n_in; (void)out_size; (void)ws_size;
  const float* f2d = (const float*)d_in[0];
  const float* f3d = (const float*)d_in[1];
  const float* Wq  = (const float*)d_in[2];
  const float* bq  = (const float*)d_in[3];
  const float* Wk  = (const float*)d_in[4];
  const float* bk  = (const float*)d_in[5];
  const float* Wv  = (const float*)d_in[6];
  const float* bv  = (const float*)d_in[7];
  const float* Wo  = (const float*)d_in[8];
  const float* bo  = (const float*)d_in[9];
  const float* Wps = (const float*)d_in[10];
  const float* bps = (const float*)d_in[11];
  float* Y = (float*)d_out;

  _Float16* ws = (_Float16*)d_ws;
  const size_t NQ = (size_t)8 * 4096 * 128;   // per-tensor f16 elements
  _Float16* Qh  = ws;            // [B][4096][128]
  _Float16* Kh  = Qh + NQ;       // [B][4096][128]  (= K^T rows)
  _Float16* VhT = Kh + NQ;       // [B][128][4096]  (transposed V)
  _Float16* OT  = VhT + NQ;      // [B][128][4096]  attention output, k-major
  _Float16* Xh  = OT + NQ;       // [B][256][4096]  residual-added x (f16)

  proj_kernel<0><<<dim3(32, 2, 8), 256, 0, stream>>>(f2d, Wq, bq, Qh, 256);
  proj_kernel<0><<<dim3(32, 2, 8), 256, 0, stream>>>(f3d, Wk, bk, Kh, 512);
  proj_kernel<1><<<dim3(32, 2, 8), 256, 0, stream>>>(f3d, Wv, bv, VhT, 512);
  attn_kernel<<<dim3(32, 8), 256, 0, stream>>>(Qh, Kh, VhT, OT);
  wo_kernel<<<dim3(32, 4, 8), 256, 0, stream>>>(OT, Wo, bo, f2d, Xh);
  conv_kernel<<<dim3(32, 4, 8), 256, 0, stream>>>(Xh, Wps, bps, Y);
}
